// CrossAttention_30880814859066
// MI455X (gfx1250) — matrix operand order repaired and checked by reading, not compiled, not run on hardware
//
#include <hip/hip_runtime.h>

#define BATCH 2
#define NCTX 1024
#define LCTX 4096
#define CDIM 1024
#define NHEADS 16
#define HDIM 64
#define ATTN_SCALE 0.125f   // 64^-0.5

typedef __attribute__((ext_vector_type(16))) __bf16 v16bf;
typedef __attribute__((ext_vector_type(8)))  __bf16 v8bf;
typedef __attribute__((ext_vector_type(8)))  float  v8f;
typedef __attribute__((ext_vector_type(4)))  float  v4f;
typedef __attribute__((ext_vector_type(4)))  unsigned int v4u;
typedef __attribute__((ext_vector_type(8)))  int    v8i;
typedef __attribute__((ext_vector_type(4)))  int    v4i;

__device__ __forceinline__ void wave_lds_sync() {
    asm volatile("s_wait_dscnt 0" ::: "memory");
}

__device__ __forceinline__ unsigned lds_off(const void* p) {
    // low 32 bits of the generic LDS address = LDS byte offset (ISA 10.2)
    return (unsigned)(unsigned long long)p;
}

// Tensor Data Mover: 2-D tile (bf16, data_size=2B) from global -> LDS.
// tile_d0 = contiguous elements per row, tile_d1 = rows,
// stride_d0 = row pitch in elements. Descriptor per CDNA5 ISA.
__device__ __forceinline__ void tdm_load_2d_bf16(unsigned ldsAddr, const void* gptr,
                                                 unsigned tensor_d0, unsigned tensor_d1,
                                                 unsigned tile_d0, unsigned tile_d1,
                                                 unsigned stride_d0) {
    unsigned long long ga = (unsigned long long)gptr;
    v4u g0;
    g0[0] = 1u;                                                  // count=1 (valid)
    g0[1] = ldsAddr;                                             // lds_addr
    g0[2] = (unsigned)(ga & 0xFFFFFFFFu);                        // global_addr[31:0]
    g0[3] = (unsigned)((ga >> 32) & 0x01FFFFFFu) | (2u << 30);   // ga[56:32] | type=2
    v8i g1;
    g1[0] = (int)(1u << 16);                                     // wg_mask=0, data_size=1 (2B)
    g1[1] = (int)((tensor_d0 & 0xFFFFu) << 16);                  // tensor_dim0[15:0]
    g1[2] = (int)(((tensor_d0 >> 16) & 0xFFFFu) | ((tensor_d1 & 0xFFFFu) << 16));
    g1[3] = (int)(((tensor_d1 >> 16) & 0xFFFFu) | ((tile_d0 & 0xFFFFu) << 16));
    g1[4] = (int)(tile_d1 & 0xFFFFu);                            // tile_dim1, tile_dim2=0
    g1[5] = (int)stride_d0;                                      // tensor_dim0_stride[31:0]
    g1[6] = 0;
    g1[7] = 0;
    v4i gz4 = {0, 0, 0, 0};
    v8i gz8 = {0, 0, 0, 0, 0, 0, 0, 0};
    __builtin_amdgcn_tensor_load_to_lds(g0, g1, gz4, gz4, gz8, 0);
}

__device__ __forceinline__ v16bf cat16(v8bf lo, v8bf hi) {
    v16bf r;
#pragma unroll
    for (int i = 0; i < 8; ++i) { r[i] = lo[i]; r[i + 8] = hi[i]; }
    return r;
}

__device__ __forceinline__ v16bf load_b_frag_lds(const __bf16* p) {
    const v8bf* bp = (const v8bf*)p;
    return cat16(bp[0], bp[2]);
}

__device__ __forceinline__ v8f zero8() {
    v8f r;
#pragma unroll
    for (int i = 0; i < 8; ++i) r[i] = 0.0f;
    return r;
}

// 16x32 bf16 A-fragment from an fp32 row: e<8 -> row[k+kb+e],
// e>=8 -> row[k+16+kb+(e-8)], kb = (lane<16)?0:8.
__device__ __forceinline__ v16bf load_a_frag_f32(const float* __restrict__ row, int k, int kb) {
    const v4f* p0 = (const v4f*)(row + k + kb);
    const v4f* p1 = (const v4f*)(row + k + 16 + kb);
    v4f f0 = p0[0], f1 = p0[1], f2 = p1[0], f3 = p1[1];
    v16bf a;
#pragma unroll
    for (int i = 0; i < 4; ++i) {
        a[i]      = (__bf16)f0[i];
        a[i + 4]  = (__bf16)f1[i];
        a[i + 8]  = (__bf16)f2[i];
        a[i + 12] = (__bf16)f3[i];
    }
    return a;
}

// ---------------- Kernel 0: fp32 -> bf16 convert ----------------
__global__ void cvt_f32_to_bf16(const float* __restrict__ src, __bf16* __restrict__ dst, int n4) {
    int i = blockIdx.x * blockDim.x + threadIdx.x;
    int stride = gridDim.x * blockDim.x;
    for (; i < n4; i += stride) {
        v4f f = ((const v4f*)src)[i];
        __bf16* d = dst + 4 * (size_t)i;
#pragma unroll
        for (int j = 0; j < 4; ++j) d[j] = (__bf16)f[j];
    }
}

// ---------------- GEMM: Out[m,n] = sum_c In[m,c] * Wb[n,c] ----------------
// In fp32 [M,C] (cast on the fly), Wb bf16 [Ntot,C].
// Weight tile (64 cols x 32 k, 4KB) staged to LDS by TDM, double buffered,
// shared by all 8 waves. Block tile 128(M) x 64(N); wave tile 16 x 64.
// MODE 0: bf16 row-major out; MODE 1: bf16 out transposed per head (V);
// MODE 2: fp32 out + bias.
template <int MODE>
__global__ void gemm_xWT(const float* __restrict__ In, const __bf16* __restrict__ Wb,
                         void* __restrict__ Out, const float* __restrict__ bias,
                         int C, int Ntot, int Lrows) {
    __shared__ __bf16 Wt[2][64 * 32];

    const int lane  = threadIdx.x & 31;
    const int wave  = threadIdx.x >> 5;
    const int m0    = blockIdx.x * 128 + wave * 16;
    const int n0    = blockIdx.y * 64;
    const int lm    = lane & 15;
    const int kb    = (lane < 16) ? 0 : 8;
    const int kb2   = (lane < 16) ? 0 : 16;
    const int rbase = (lane < 16) ? 0 : 8;

    const float* arow = In + (size_t)(m0 + lm) * C;
    const __bf16* wtile0 = Wb + (size_t)n0 * C;

    v8f acc[4];
#pragma unroll
    for (int t = 0; t < 4; ++t) acc[t] = zero8();

    const int nk = C / 32;
    if (threadIdx.x == 0)
        tdm_load_2d_bf16(lds_off(&Wt[0][0]), wtile0, C, Ntot, 32, 64, C);

    for (int i = 0; i < nk; ++i) {
        const int k = i * 32;
        const int cur = i & 1;
        if (threadIdx.x == 0) {
            if (i + 1 < nk) {
                tdm_load_2d_bf16(lds_off(&Wt[cur ^ 1][0]), wtile0 + (k + 32), C, Ntot, 32, 64, C);
                __builtin_amdgcn_s_wait_tensorcnt(1);
            } else {
                __builtin_amdgcn_s_wait_tensorcnt(0);
            }
        }
        __syncthreads();

        v16bf a = load_a_frag_f32(arow, k, kb);
#pragma unroll
        for (int t = 0; t < 4; ++t) {
            v16bf b = load_b_frag_lds(&Wt[cur][(t * 16 + lm) * 32 + kb]);
            acc[t] = __builtin_amdgcn_wmma_f32_16x16x32_bf16(
                false, a, false, b, (short)0, acc[t], false, false);
        }
        __syncthreads();   // all waves done with Wt[cur] before it is overwritten
    }

#pragma unroll
    for (int t = 0; t < 4; ++t) {
        const int ncol = n0 + t * 16 + lm;
        if (MODE == 2) {
            float bv = bias[ncol];
            float* o = (float*)Out;
#pragma unroll
            for (int r = 0; r < 8; ++r)
                o[(size_t)(m0 + rbase + r) * Ntot + ncol] = acc[t][r] + bv;
        } else if (MODE == 0) {
            __bf16* o = (__bf16*)Out;
#pragma unroll
            for (int r = 0; r < 8; ++r)
                o[(size_t)(m0 + rbase + r) * Ntot + ncol] = (__bf16)acc[t][r];
        } else {  // MODE 1: V transposed per head -> [B,H,D,L]
            const int h = ncol >> 6, d = ncol & 63;
            __bf16* o = (__bf16*)Out;
#pragma unroll
            for (int r = 0; r < 8; ++r) {
                const int m = m0 + rbase + r;
                const int b = m / Lrows;
                const int l = m - b * Lrows;
                o[((size_t)((b * NHEADS + h) * HDIM + d)) * Lrows + l] = (__bf16)acc[t][r];
            }
        }
    }
}

// ---------------- Attention ----------------
// One wave per (b,h,16 q-rows); 8 waves per block share one (b,h), so the
// 32x64 K tile and 64x32 V^T tile are TDM-staged to LDS once per block,
// double buffered, overlapped with WMMA via s_wait_tensorcnt.
__global__ void attn_kernel(const __bf16* __restrict__ qw, const __bf16* __restrict__ kw,
                            const __bf16* __restrict__ vT, float* __restrict__ attn_out) {
    __shared__ __bf16 Kt[2][32 * 64];   // [l within tile][d]  (8KB)
    __shared__ __bf16 Vt[2][64 * 32];   // [d within head][l]  (8KB)
    __shared__ float  lds_S[8][16 * 32];
    __shared__ __bf16 lds_P[8][16 * 32];
    __shared__ float  lds_A[8][16];

    const int lane = threadIdx.x & 31;
    const int wave = threadIdx.x >> 5;
    const int task = blockIdx.x * 8 + wave;   // 8 consecutive q-tiles, same (b,h)
    const int qt = task & 63;
    const int h  = (task >> 6) & 15;
    const int b  = task >> 10;
    const int n0 = qt * 16;

    float*  S  = lds_S[wave];
    __bf16* P  = lds_P[wave];
    float*  AL = lds_A[wave];

    const int lm    = lane & 15;
    const int kb    = (lane < 16) ? 0 : 8;
    const int kb2   = (lane < 16) ? 0 : 16;
    const int rbase = (lane < 16) ? 0 : 8;

    // Q fragments: row n0+lm of head h, two 32-wide chunks over D=64
    const __bf16* qrow = qw + (size_t)(b * NCTX + n0 + lm) * CDIM + h * HDIM;
    v16bf qA[2];
#pragma unroll
    for (int kc = 0; kc < 2; ++kc) {
        const v8bf* plo = (const v8bf*)(qrow + kc * 32 + kb);
        const v8bf* phi = (const v8bf*)(qrow + kc * 32 + 16 + kb);
        qA[kc] = cat16(plo[0], phi[0]);
    }

    const __bf16* kbase = kw + (size_t)(b * LCTX) * CDIM + h * HDIM;               // +l*CDIM
    const __bf16* vbase = vT + (size_t)((b * NHEADS + h) * HDIM) * LCTX;           // +d*LCTX

    float m_i = -INFINITY;
    float l_i = 0.0f;
    v8f O[4];
#pragma unroll
    for (int t = 0; t < 4; ++t) O[t] = zero8();

    const int NITER = LCTX / 32;   // 128
    if (threadIdx.x == 0) {
        tdm_load_2d_bf16(lds_off(&Kt[0][0]), kbase, CDIM, BATCH * LCTX, 64, 32, CDIM);
        tdm_load_2d_bf16(lds_off(&Vt[0][0]), vbase, LCTX, BATCH * NHEADS * HDIM, 32, 64, LCTX);
    }

    for (int i = 0; i < NITER; ++i) {
        const int l0 = i * 32;
        const int cur = i & 1;
        if (threadIdx.x == 0) {
            if (i + 1 < NITER) {
                tdm_load_2d_bf16(lds_off(&Kt[cur ^ 1][0]), kbase + (size_t)(l0 + 32) * CDIM,
                                 CDIM, BATCH * LCTX, 64, 32, CDIM);
                tdm_load_2d_bf16(lds_off(&Vt[cur ^ 1][0]), vbase + (l0 + 32),
                                 LCTX, BATCH * NHEADS * HDIM, 32, 64, LCTX);
                __builtin_amdgcn_s_wait_tensorcnt(2);
            } else {
                __builtin_amdgcn_s_wait_tensorcnt(0);
            }
        }
        __syncthreads();
        const __bf16* Kc = Kt[cur];
        const __bf16* Vc = Vt[cur];

        // ---- scores S[16 x 32] = scale * Q K^T ------------------------------
#pragma unroll
        for (int j = 0; j < 2; ++j) {
            v8f s = zero8();
            {
                v16bf bb = load_b_frag_lds(Kc + (j * 16 + lm) * 64 + kb);
                s = __builtin_amdgcn_wmma_f32_16x16x32_bf16(false, qA[0], false, bb, (short)0, s, false, false);
            }
            {
                v16bf bb = load_b_frag_lds(Kc + (j * 16 + lm) * 64 + 32 + kb);
                s = __builtin_amdgcn_wmma_f32_16x16x32_bf16(false, qA[1], false, bb, (short)0, s, false, false);
            }
#pragma unroll
            for (int r = 0; r < 8; ++r)
                S[(rbase + r) * 32 + j * 16 + lm] = s[r] * ATTN_SCALE;
        }
        wave_lds_sync();

        // ---- online softmax (lanes 0..15 each own one q-row) ----------------
        if (lane < 16) {
            const float* srow = S + lane * 32;
            float tmax = -INFINITY;
            float vals[32];
#pragma unroll
            for (int ii = 0; ii < 32; ++ii) { vals[ii] = srow[ii]; tmax = fmaxf(tmax, vals[ii]); }
            const float m_new = fmaxf(m_i, tmax);
            float tsum = 0.0f;
#pragma unroll
            for (int ii = 0; ii < 32; ++ii) {
                float p = __expf(vals[ii] - m_new);
                tsum += p;
                P[lane * 32 + ii] = (__bf16)p;
            }
            const float alpha = __expf(m_i - m_new);
            l_i = l_i * alpha + tsum;
            m_i = m_new;
            AL[lane] = alpha;
        }
        wave_lds_sync();

        // ---- rescale O and accumulate P @ V ---------------------------------
        float ar[8];
#pragma unroll
        for (int r = 0; r < 8; ++r) ar[r] = AL[rbase + r];

        v16bf pA;
        {
            const v8bf* plo = (const v8bf*)(P + lm * 32 + kb);
            const v8bf* phi = (const v8bf*)(P + lm * 32 + 16 + kb);
            pA = cat16(plo[0], phi[0]);
        }
#pragma unroll
        for (int t = 0; t < 4; ++t) {
#pragma unroll
            for (int r = 0; r < 8; ++r) O[t][r] *= ar[r];
            v16bf vb = load_b_frag_lds(Vc + (t * 16 + lm) * 32 + kb);
            O[t] = __builtin_amdgcn_wmma_f32_16x16x32_bf16(false, pA, false, vb, (short)0, O[t], false, false);
        }
        __syncthreads();   // Kt/Vt[cur] free for next TDM; S/P/AL WAR safe
    }

    // ---- normalize by row sums and store fp32 -------------------------------
    if (lane < 16) AL[lane] = 1.0f / l_i;
    wave_lds_sync();
    float lr[8];
#pragma unroll
    for (int r = 0; r < 8; ++r) lr[r] = AL[rbase + r];
#pragma unroll
    for (int t = 0; t < 4; ++t) {
#pragma unroll
        for (int r = 0; r < 8; ++r) {
            attn_out[(size_t)(b * NCTX + n0 + rbase + r) * CDIM + h * HDIM + t * 16 + lm] =
                O[t][r] * lr[r];
        }
    }
}

// ---------------- Host launcher ----------------
extern "C" void kernel_launch(void* const* d_in, const int* in_sizes, int n_in,
                              void* d_out, int out_size, void* d_ws, size_t ws_size,
                              hipStream_t stream) {
    const float* x  = (const float*)d_in[0];
    const float* y  = (const float*)d_in[1];
    const float* Wq = (const float*)d_in[2];
    const float* Wk = (const float*)d_in[3];
    const float* Wv = (const float*)d_in[4];
    const float* Wp = (const float*)d_in[5];
    const float* bp = (const float*)d_in[6];

    char* ws = (char*)d_ws;
    const size_t WBYTES = (size_t)CDIM * CDIM * sizeof(__bf16);  // 2 MB
    __bf16* Wq_b = (__bf16*)(ws + 0 * WBYTES);
    __bf16* Wk_b = (__bf16*)(ws + 1 * WBYTES);
    __bf16* Wv_b = (__bf16*)(ws + 2 * WBYTES);
    __bf16* Wp_b = (__bf16*)(ws + 3 * WBYTES);
    size_t off = 4 * WBYTES;
    __bf16* q_b  = (__bf16*)(ws + off); off += (size_t)BATCH * NCTX * CDIM * sizeof(__bf16);
    __bf16* k_b  = (__bf16*)(ws + off); off += (size_t)BATCH * LCTX * CDIM * sizeof(__bf16);
    __bf16* vT_b = (__bf16*)(ws + off); off += (size_t)BATCH * LCTX * CDIM * sizeof(__bf16);
    float*  at_f = (float*)(ws + off);

    const int nW4 = CDIM * CDIM / 4;
    cvt_f32_to_bf16<<<1024, 256, 0, stream>>>(Wq, Wq_b, nW4);
    cvt_f32_to_bf16<<<1024, 256, 0, stream>>>(Wk, Wk_b, nW4);
    cvt_f32_to_bf16<<<1024, 256, 0, stream>>>(Wv, Wv_b, nW4);
    cvt_f32_to_bf16<<<1024, 256, 0, stream>>>(Wp, Wp_b, nW4);

    gemm_xWT<0><<<dim3(16, 16), 256, 0, stream>>>(x, Wq_b, (void*)q_b, nullptr, CDIM, CDIM, 0);
    gemm_xWT<0><<<dim3(64, 16), 256, 0, stream>>>(y, Wk_b, (void*)k_b, nullptr, CDIM, CDIM, 0);
    gemm_xWT<1><<<dim3(64, 16), 256, 0, stream>>>(y, Wv_b, (void*)vT_b, nullptr, CDIM, CDIM, LCTX);

    attn_kernel<<<256, 256, 0, stream>>>(q_b, k_b, vT_b, at_f);

    gemm_xWT<2><<<dim3(16, 16), 256, 0, stream>>>(at_f, Wp_b, d_out, bp, CDIM, CDIM, 0);
}
